// SCL_83150566851170
// MI455X (gfx1250) — matrix operand order repaired and checked by reading, not compiled, not run on hardware
//
#include <hip/hip_runtime.h>

typedef __attribute__((ext_vector_type(16))) _Float16 v16h;
typedef __attribute__((ext_vector_type(8)))  _Float16 v8h;
typedef __attribute__((ext_vector_type(4)))  _Float16 v4h;
typedef __attribute__((ext_vector_type(2)))  __fp16   v2hf;
typedef __attribute__((ext_vector_type(8)))  float    v8f;

namespace {
constexpr int NN = 64;     // nodes
constexpr int HH = 128;    // hidden
constexpr int LL = 256;
constexpr int BS = 32;
constexpr int NB = BS * LL;   // 8192 graph instances

// dynamic LDS byte offsets
constexpr int OFF_A   = 0;                // 16KB A fp32 (then A_hat fp32)
constexpr int OFF_AH  = 16384;            // 8KB  A_hat f16; reused as sRed [64][16] f32 in matmul2
constexpr int OFF_H1T = OFF_AH + 8192;    // 16KB h1 transposed [128][64] f16
constexpr int OFF_C1  = OFF_H1T + 16384;  // 16KB C1 = A_hat@h1 [64][128] f16
constexpr int OFF_W2  = OFF_C1 + 16384;   // 32KB W2 f16 [o][f]
constexpr int OFF_D   = OFF_W2 + 32768;   // 256B d (then ax)
constexpr int OFF_B2  = OFF_D  + 256;
constexpr int OFF_W1  = OFF_B2 + 512;
constexpr int OFF_B1  = OFF_W1 + 512;
constexpr int OFF_WC  = OFF_B1 + 512;
constexpr int OFF_X   = OFF_WC + 512;
constexpr int SMEM_BYTES = OFF_X + 256;   // 92672 B
}

__launch_bounds__(128)
__global__ void gcn_fused_kernel(const float* __restrict__ X,  const float* __restrict__ A,
                                 const float* __restrict__ W1, const float* __restrict__ b1,
                                 const float* __restrict__ W2, const float* __restrict__ b2,
                                 const float* __restrict__ wc, const float* __restrict__ bc,
                                 float* __restrict__ out)
{
  extern __shared__ char smem[];
  float*    sA   = (float*)(smem + OFF_A);       // [64][64] fp32
  _Float16* sAh  = (_Float16*)(smem + OFF_AH);   // [64][64] f16
  float*    sRed = (float*)(smem + OFF_AH);      // [64][16] f32 (after matmul1 barrier)
  _Float16* sH1T = (_Float16*)(smem + OFF_H1T);  // [128][64] f16: h1^T
  _Float16* sC1  = (_Float16*)(smem + OFF_C1);   // [64][128] f16
  _Float16* sW2  = (_Float16*)(smem + OFF_W2);   // [128][128] f16, row-major [o][f]
  float*    sd   = (float*)(smem + OFF_D);       // d, then ax
  float*    sB2  = (float*)(smem + OFF_B2);
  float*    sW1  = (float*)(smem + OFF_W1);
  float*    sB1  = (float*)(smem + OFF_B1);
  float*    sWC  = (float*)(smem + OFF_WC);
  float*    sx   = (float*)(smem + OFF_X);

  const int t     = threadIdx.x;
  const int lane  = t & 31;
  const int wave  = t >> 5;
  const int b     = blockIdx.x;        // graph instance = bsIdx*L + l
  const int bsIdx = b >> 8;
  const int l     = b & 255;

  // ---- stage small params + x column ----
  if (t < HH) { sW1[t] = W1[t]; sB1[t] = b1[t]; sB2[t] = b2[t]; sWC[t] = wc[t]; }
  if (t < NN) sx[t] = X[(size_t)bsIdx * NN * LL + (size_t)t * LL + l];

  // ---- W2 fp32 -> f16 in LDS (L2-resident across blocks), packed b128 stores ----
  {
    const float4* w4 = (const float4*)W2;
    #pragma unroll
    for (int i = 0; i < 16; ++i) {
      float4 a = w4[t * 32 + 2 * i];
      float4 c = w4[t * 32 + 2 * i + 1];
      v8h pk = { (_Float16)a.x, (_Float16)a.y, (_Float16)a.z, (_Float16)a.w,
                 (_Float16)c.x, (_Float16)c.y, (_Float16)c.z, (_Float16)c.w };
      *(v8h*)(sW2 + t * 128 + i * 8) = pk;
    }
  }

  // ---- load A tile, add self-loops ----
  {
    const float4* a4 = (const float4*)(A + (size_t)b * NN * NN);
    #pragma unroll
    for (int i = 0; i < 8; ++i) {
      float4 v = a4[t * 8 + i];
      int o  = (t * 8 + i) * 4;
      int r0 = o >> 6, c0 = o & 63;
      if (r0 == c0    ) v.x += 1.f;
      if (r0 == c0 + 1) v.y += 1.f;
      if (r0 == c0 + 2) v.z += 1.f;
      if (r0 == c0 + 3) v.w += 1.f;
      ((float4*)sA)[t * 8 + i] = v;
    }
  }
  __syncthreads();

  // ---- d = rowsum^(-1/2) (float4 chunks, staggered) ----
  if (t < NN) {
    const float4* row4 = (const float4*)(sA + t * NN);
    float s = 0.f;
    #pragma unroll
    for (int c0 = 0; c0 < 16; ++c0) {
      int c = (c0 + t) & 15;
      float4 v = row4[c];
      s += (v.x + v.y) + (v.z + v.w);
    }
    sd[t] = rsqrtf(s);
  }
  __syncthreads();

  // ---- A_hat = d_i * A * d_j : fp32 in place + f16 copy (2 threads per row) ----
  {
    const int r     = t >> 1;
    const int cbase = (t & 1) * 32;
    const float dr  = sd[r];
    #pragma unroll
    for (int q0 = 0; q0 < 8; ++q0) {
      int q = (q0 + r) & 7;
      int c = cbase + q * 4;
      float4 av = *(const float4*)(sA + r * NN + c);
      float4 dv = *(const float4*)(sd + c);
      float4 hv;
      hv.x = dr * av.x * dv.x; hv.y = dr * av.y * dv.y;
      hv.z = dr * av.z * dv.z; hv.w = dr * av.w * dv.w;
      *(float4*)(sA + r * NN + c) = hv;
      v4h pk = { (_Float16)hv.x, (_Float16)hv.y, (_Float16)hv.z, (_Float16)hv.w };
      *(v4h*)(sAh + r * NN + c) = pk;
    }
  }
  __syncthreads();

  // ---- ax = A_hat @ x (fp32 matvec); store over sd ----
  if (t < NN) {
    const float4* row4 = (const float4*)(sA + t * NN);
    const float4* x4   = (const float4*)sx;
    float s = 0.f;
    #pragma unroll
    for (int c0 = 0; c0 < 16; ++c0) {
      int c = (c0 + t) & 15;
      float4 v = row4[c];
      float4 xv = x4[c];
      s += v.x * xv.x + v.y * xv.y + v.z * xv.z + v.w * xv.w;
    }
    sd[t] = s;
  }
  __syncthreads();

  // ---- h1^T[f][j] = relu(ax[j]*W1[f] + b1[f]) -> f16, b128 stores ----
  {
    const float w1f = sW1[t], b1f = sB1[t];
    const float4* d4 = (const float4*)sd;
    _Float16* rowp = sH1T + t * NN;
    #pragma unroll
    for (int c0 = 0; c0 < 8; ++c0) {
      int cc = (c0 + t) & 7;             // 8-half chunk index
      float4 a = d4[2 * cc];
      float4 c = d4[2 * cc + 1];
      v8h pk = { (_Float16)fmaxf(a.x * w1f + b1f, 0.f), (_Float16)fmaxf(a.y * w1f + b1f, 0.f),
                 (_Float16)fmaxf(a.z * w1f + b1f, 0.f), (_Float16)fmaxf(a.w * w1f + b1f, 0.f),
                 (_Float16)fmaxf(c.x * w1f + b1f, 0.f), (_Float16)fmaxf(c.y * w1f + b1f, 0.f),
                 (_Float16)fmaxf(c.z * w1f + b1f, 0.f), (_Float16)fmaxf(c.w * w1f + b1f, 0.f) };
      *(v8h*)(rowp + cc * 8) = pk;
    }
  }
  __syncthreads();

  // ---- fragment helpers (wave32 WMMA 16x16x32 f16 layouts) ----
  const int m0   = wave * 16;
  const int lrow = lane & 15;
  const int lhi  = (lane >> 4) & 1;

  auto loadA = [](const _Float16* base, int k0) {   // A-frag: 16 contiguous halfs
    v16h f;
    #pragma unroll
    for (int h = 0; h < 16; ++h) f[h] = base[k0 + h];
    return f;
  };
  auto loadB = [](const _Float16* base, int off) {  // B-frag: 16 contiguous halfs
    v16h f;
    #pragma unroll
    for (int h = 0; h < 16; ++h) f[h] = base[off + h];
    return f;
  };

  // ---- matmul1: C1 = A_hat(64x64) @ H1(64x128); A-frags hoisted, B loads batched ----
  {
    const _Float16* aBase = sAh  + (m0 + lrow) * NN + lhi * 16;
    const _Float16* bBase = sH1T + lrow * NN + lhi * 16;
    _Float16*       cBase = sC1  + (m0 + lhi * 8) * HH + lrow;

    const v16h af0 = loadA(aBase, 0);
    const v16h af1 = loadA(aBase, 32);

    #pragma unroll
    for (int nt = 0; nt < 8; ++nt) {
      // batch all B loads first -> partial s_wait_dscnt before each wmma
      v16h bf0 = loadB(bBase, nt * 16 * NN);
      v16h bf1 = loadB(bBase, nt * 16 * NN + 32);
      v8f acc = {};
      acc = __builtin_amdgcn_wmma_f32_16x16x32_f16(false, af0, false, bf0,
                                                   (short)0, acc, false, false);
      acc = __builtin_amdgcn_wmma_f32_16x16x32_f16(false, af1, false, bf1,
                                                   (short)0, acc, false, false);
      #pragma unroll
      for (int v = 0; v < 8; v += 2) {
        v2hf pk = __builtin_amdgcn_cvt_pkrtz(acc[v], acc[v + 1]);
        cBase[(v    ) * HH + nt * 16] = (_Float16)pk[0];
        cBase[(v + 1) * HH + nt * 16] = (_Float16)pk[1];
      }
    }
  }
  __syncthreads();   // required: sRed (below) aliases sAh rows read by other waves

  // ---- matmul2: C2 = C1(64x128) @ W2^T; fused relu+bias+wc reduction in registers ----
  {
    const _Float16* aBase = sC1 + (m0 + lrow) * HH + lhi * 16;
    const _Float16* bBase = sW2 + lrow * HH + lhi * 16;
    const float*    b2p   = sB2 + lrow;
    const float*    wcp   = sWC + lrow;

    const v16h ar0 = loadA(aBase, 0);
    const v16h ar1 = loadA(aBase, 32);
    const v16h ar2 = loadA(aBase, 64);
    const v16h ar3 = loadA(aBase, 96);

    float preg[8];
    #pragma unroll
    for (int v = 0; v < 8; ++v) preg[v] = 0.f;

    #pragma unroll
    for (int nt = 0; nt < 8; ++nt) {
      // batch all 8 b128 B loads, then run the 4-wmma chain on partial waits
      v16h bf0 = loadB(bBase, nt * 16 * HH);
      v16h bf1 = loadB(bBase, nt * 16 * HH + 32);
      v16h bf2 = loadB(bBase, nt * 16 * HH + 64);
      v16h bf3 = loadB(bBase, nt * 16 * HH + 96);
      v8f acc = {};
      acc = __builtin_amdgcn_wmma_f32_16x16x32_f16(false, ar0, false, bf0,
                                                   (short)0, acc, false, false);
      acc = __builtin_amdgcn_wmma_f32_16x16x32_f16(false, ar1, false, bf1,
                                                   (short)0, acc, false, false);
      acc = __builtin_amdgcn_wmma_f32_16x16x32_f16(false, ar2, false, bf2,
                                                   (short)0, acc, false, false);
      acc = __builtin_amdgcn_wmma_f32_16x16x32_f16(false, ar3, false, bf3,
                                                   (short)0, acc, false, false);
      const float b2c = b2p[nt * 16];
      const float wcc = wcp[nt * 16];
      #pragma unroll
      for (int v = 0; v < 8; ++v)
        preg[v] += fmaxf(acc[v] + b2c, 0.f) * wcc;   // h2[row][col] * wc[col]
    }
    // lane-partial per row: rows rbase..rbase+7, 16 column-groups across lanes
    float* redBase = sRed + (m0 + lhi * 8) * 16 + lrow;
    #pragma unroll
    for (int v = 0; v < 8; ++v)
      redBase[v * 16] = preg[v];
  }
  __syncthreads();

  // ---- out[n] = sum over 16 lane-partials + bc ----
  if (t < NN) {
    const float4* r4 = (const float4*)(sRed + t * 16);
    float4 p0 = r4[0], p1 = r4[1], p2 = r4[2], p3 = r4[3];
    float s = ((p0.x + p0.y) + (p0.z + p0.w)) + ((p1.x + p1.y) + (p1.z + p1.w))
            + ((p2.x + p2.y) + (p2.z + p2.w)) + ((p3.x + p3.y) + (p3.z + p3.w));
    out[(size_t)bsIdx * NN * LL + (size_t)t * LL + l] = s + bc[0];
  }
}

extern "C" void kernel_launch(void* const* d_in, const int* in_sizes, int n_in,
                              void* d_out, int out_size, void* d_ws, size_t ws_size,
                              hipStream_t stream) {
  const float* X  = (const float*)d_in[0];
  const float* A  = (const float*)d_in[1];
  const float* W1 = (const float*)d_in[2];
  const float* b1 = (const float*)d_in[3];
  const float* W2 = (const float*)d_in[4];
  const float* b2 = (const float*)d_in[5];
  const float* wc = (const float*)d_in[6];
  const float* bc = (const float*)d_in[7];
  float* out = (float*)d_out;

  hipLaunchKernelGGL(gcn_fused_kernel, dim3(NB), dim3(128), SMEM_BYTES, stream,
                     X, A, W1, b1, W2, b2, wc, bc, out);
}
